// GaussianHeadModule_51041391345917
// MI455X (gfx1250) — compile-verified, matrix operand order checked by reading, not run on hardware
//
#include <hip/hip_runtime.h>
#include <hip/hip_bf16.h>
#include <math.h>

#define DEVINL __device__ __forceinline__

typedef __attribute__((ext_vector_type(16))) _Float16 v16h;
typedef __attribute__((ext_vector_type(8)))  float    v8f;

constexpr int NPTS  = 100000;
constexpr int BATCH = 2;
constexpr int TM    = 128;                       // points per tile
constexpr int TILES = (NPTS + TM - 1) / TM;      // 782
constexpr int ASTRIDE = 264;                     // activation row stride (f16): 256 + 8 pad (16 distinct banks across 16 rows)
constexpr int FSTRIDE = 136;                     // tanh(feature) row stride (f16)

// ---- d_out layout (floats, concatenated return order: xyz,color,scales,rot,opacity)
constexpr size_t OUT_XYZ    = 0;
constexpr size_t OUT_COLOR  = (size_t)BATCH * NPTS * 3;
constexpr size_t OUT_SCALES = OUT_COLOR  + (size_t)BATCH * NPTS * 32;
constexpr size_t OUT_ROT    = OUT_SCALES + (size_t)BATCH * NPTS * 3;
constexpr size_t OUT_OP     = OUT_ROT    + (size_t)BATCH * NPTS * 4;

// ---- LDS layout (bytes)
constexpr size_t SH_BUFA  = 0;
constexpr size_t SH_BUFB  = SH_BUFA  + (size_t)TM * ASTRIDE * 2;   // 67584
constexpr size_t SH_FEAT  = SH_BUFB  + (size_t)TM * ASTRIDE * 2;   // 135168
constexpr size_t SH_XPE   = SH_FEAT  + (size_t)TM * FSTRIDE * 2;   // 169984
constexpr size_t SH_COLOR = SH_XPE   + (size_t)TM * 32 * 2;        // 178176
constexpr size_t SH_DATTR = SH_COLOR + (size_t)TM * 36 * 4;        // 196608
constexpr size_t SH_DXYZ  = SH_DATTR + (size_t)TM * 9  * 4;        // 201216
constexpr size_t SH_EW    = SH_DXYZ  + (size_t)TM * 4  * 4;        // 203264
constexpr size_t SH_XYZ   = SH_EW    + (size_t)TM * 4;             // 203776
constexpr size_t SH_LM    = SH_XYZ   + (size_t)TM * 3 * 4;         // 205312
constexpr size_t SH_VEC   = SH_LM    + 832;                        // 206144 (expv 64 f16 | posev 64 f16)
constexpr size_t SH_RTS   = SH_VEC   + 256;                        // 206400 (R9,T3,S1 f32)
constexpr size_t SH_TOTAL = SH_RTS   + 64;                         // 206464 bytes (< 320KB WGP LDS)

// ---------------------------------------------------------------------------
// WMMA fragment helpers (16x16x32 f16, f32 accum).
// A (16x32, M x K): row m = lane&15 lives in one lane; K pairs per VGPR:
//   lanes 0-15 : v0..3 -> K=2v..2v+1, v4..7 -> K=2v+8..
//   lanes 16-31: v0..3 -> K=2v+8..,   v4..7 -> K=2v+16..
// (u32 index offsets a4 = grp?4:0, b4 = grp?8:4)
// B is pre-packed by gh_pack into per-lane-contiguous 32B chunks so each lane
// loads its whole fragment with two global_load_b128.
// ---------------------------------------------------------------------------
union Frag16 { v16h h; unsigned u[8]; };

DEVINL v16h load_a(const _Float16* arow, int kt, int a4, int b4) {
  const unsigned* p = (const unsigned*)(arow + kt * 32);
  Frag16 f;
#pragma unroll
  for (int v = 0; v < 8; ++v) f.u[v] = p[v + (v < 4 ? a4 : b4)];
  return f.h;
}

DEVINL v16h load_b(const _Float16* wp, int frag, int lane) {
  return *(const v16h*)(wp + ((size_t)frag << 9) + (lane << 4));
}

DEVINL v8f wmma16(v16h a, v16h b, v8f c) {
  return __builtin_amdgcn_wmma_f32_16x16x32_f16(false, a, false, b, (short)0, c, false, false);
}

// Hidden layer: LDS f16 act[128][K] -> LDS f16 act[128][N], leaky ReLU.
// Wave owns rows [row0,row0+16); loops all col tiles; 2 accumulators for ILP.
template<int KP, int NP, bool LEAKY>
DEVINL void run_layer(const _Float16* __restrict__ actIn, _Float16* __restrict__ actOut,
                      const _Float16* __restrict__ wp, const float* __restrict__ bias,
                      int row0, int lane, int a4, int b4) {
  constexpr int KT = KP / 32, CT = NP / 16;
  const int grp  = lane >> 4;
  const int ncol = lane & 15;
  const _Float16* arow = actIn + (size_t)(row0 + ncol) * ASTRIDE;
  _Float16* orow = actOut + (size_t)(row0 + grp * 8) * ASTRIDE + ncol;
  for (int ct = 0; ct < CT; ++ct) {
    float bv = bias[ct * 16 + ncol];
    v8f acc  = {bv, bv, bv, bv, bv, bv, bv, bv};
    v8f acc2 = {0.f, 0.f, 0.f, 0.f, 0.f, 0.f, 0.f, 0.f};
#pragma unroll
    for (int kt = 0; kt < KT; kt += 2) {
      acc = wmma16(load_a(arow, kt, a4, b4), load_b(wp, ct * KT + kt, lane), acc);
      if (kt + 1 < KT)
        acc2 = wmma16(load_a(arow, kt + 1, a4, b4), load_b(wp, ct * KT + kt + 1, lane), acc2);
    }
#pragma unroll
    for (int r = 0; r < 8; ++r) {
      float x = acc[r] + acc2[r];
      if (LEAKY) x = x > 0.f ? x : 0.01f * x;
      orow[(size_t)r * ASTRIDE + ct * 16] = (_Float16)x;
    }
  }
}

// Final layer: -> f32 LDS buffer, optional tanh, scaled by ew (or 1-ew), = or +=.
template<int KP, int NREAL, bool TANH, bool ACCUM>
DEVINL void run_final(const _Float16* __restrict__ actIn, float* __restrict__ outBuf, int ostride,
                      const _Float16* __restrict__ wp, const float* __restrict__ bias,
                      const float* __restrict__ s_ew, bool usePw,
                      int row0, int lane, int a4, int b4) {
  constexpr int KT = KP / 32, CT = (NREAL + 15) / 16;
  const int grp  = lane >> 4;
  const int ncol = lane & 15;
  const _Float16* arow = actIn + (size_t)(row0 + ncol) * ASTRIDE;
  for (int ct = 0; ct < CT; ++ct) {
    int n = ct * 16 + ncol;
    float bv = (n < NREAL) ? bias[n] : 0.f;
    v8f acc  = {bv, bv, bv, bv, bv, bv, bv, bv};
    v8f acc2 = {0.f, 0.f, 0.f, 0.f, 0.f, 0.f, 0.f, 0.f};
#pragma unroll
    for (int kt = 0; kt < KT; kt += 2) {
      acc = wmma16(load_a(arow, kt, a4, b4), load_b(wp, ct * KT + kt, lane), acc);
      if (kt + 1 < KT)
        acc2 = wmma16(load_a(arow, kt + 1, a4, b4), load_b(wp, ct * KT + kt + 1, lane), acc2);
    }
    if (n < NREAL) {
#pragma unroll
      for (int r = 0; r < 8; ++r) {
        int m = row0 + grp * 8 + r;
        float w = s_ew[m];
        if (usePw) w = 1.f - w;
        float x = acc[r] + acc2[r];
        if (TANH) x = tanhf(x);
        x *= w;
        if (ACCUM) outBuf[m * ostride + n] += x;
        else       outBuf[m * ostride + n]  = x;
      }
    }
  }
}

// Build [tanh(feat)(128) | vec(64)] input rows (cols 182..191 of pose vec are 0).
DEVINL void build_fv(_Float16* buf, const _Float16* s_feat, const _Float16* vec,
                     int row0, int lane) {
  int r = row0 + (lane & 15);
  int h = lane >> 4;
  const unsigned* fs = (const unsigned*)(s_feat + (size_t)r * FSTRIDE);
  const unsigned* vs = (const unsigned*)vec;
  unsigned* d = (unsigned*)(buf + (size_t)r * ASTRIDE);
#pragma unroll 4
  for (int c = h; c < 64; c += 2) d[c] = fs[c];
#pragma unroll
  for (int c = h; c < 32; c += 2) d[64 + c] = vs[c];
}

// Build [posemb(xyz)(27) | vec(64) | zero-pad -> 96] input rows.
DEVINL void build_dv(_Float16* buf, const _Float16* s_xyzpe, const _Float16* vec,
                     int row0, int lane) {
  int r = row0 + (lane & 15);
  int h = lane >> 4;
  const _Float16* xp = s_xyzpe + (size_t)r * 32;
  _Float16* d = buf + (size_t)r * ASTRIDE;
  for (int c = h; c < 96; c += 2) {
    _Float16 v = (_Float16)0.f;
    if (c < 27) v = xp[c];
    else if (c - 27 < 64) v = vec[c - 27];
    d[c] = v;
  }
}

// ---------------------------------------------------------------------------
// Prep kernel 1: pack one weight matrix (f32 [NOUT][K] row-major) into f16
// WMMA-B fragment order, K padded to KP, N padded to NP (zeros).
// ---------------------------------------------------------------------------
__global__ void gh_pack(const float* __restrict__ W, _Float16* __restrict__ dst,
                        int NOUT, int K, int KP, int NP) {
  int idx = blockIdx.x * blockDim.x + threadIdx.x;   // u32 index
  int KT = KP / 32;
  int total = (NP / 16) * KT * 256;
  if (idx >= total) return;
  int v    = idx & 7;
  int lane = (idx >> 3) & 31;
  int frag = idx >> 8;
  int kt = frag % KT, ct = frag / KT;
  int grp = lane >> 4;
  int k0 = kt * 32 + 2 * v + (v < 4 ? (grp ? 8 : 0) : (grp ? 16 : 8));
  int n  = ct * 16 + (lane & 15);
  _Float16 w0 = (_Float16)0.f, w1 = (_Float16)0.f;
  if (n < NOUT) {
    if (k0 < K)     w0 = (_Float16)W[(size_t)n * K + k0];
    if (k0 + 1 < K) w1 = (_Float16)W[(size_t)n * K + k0 + 1];
  }
  union { _Float16 h[2]; unsigned u; } pk;
  pk.h[0] = w0; pk.h[1] = w1;
  ((unsigned*)dst)[idx] = pk.u;
}

// ---------------------------------------------------------------------------
// Prep kernel 2: per-batch exp f16, posemb(pose) f16 (padded to 64), so3_exp R,
// T, S into ws.
// ---------------------------------------------------------------------------
__global__ void gh_prep(const float* __restrict__ exp_coeff, const float* __restrict__ pose,
                        const float* __restrict__ scale,
                        _Float16* __restrict__ expv, _Float16* __restrict__ posev,
                        float* __restrict__ rts) {
  int t = threadIdx.x;
  if (t < BATCH * 64) expv[t] = (_Float16)exp_coeff[t];
  if (t < BATCH) {
    const float* pp = pose + t * 6;
    _Float16* pv = posev + t * 64;
    for (int j = 0; j < 6; ++j) pv[j] = (_Float16)pp[j];
    for (int i = 0; i < 4; ++i) {
      float f = (float)(1 << i);
      for (int j = 0; j < 6; ++j) {
        pv[6 + 12 * i + j]     = (_Float16)sinf(pp[j] * f);
        pv[6 + 12 * i + 6 + j] = (_Float16)cosf(pp[j] * f);
      }
    }
    for (int j = 54; j < 64; ++j) pv[j] = (_Float16)0.f;
    // so3_exp
    float wx = pp[0], wy = pp[1], wz = pp[2];
    float n2 = wx * wx + wy * wy + wz * wz;
    float th = sqrtf(fmaxf(n2, 1e-4f));
    float a  = sinf(th) / th;
    float bb = (1.f - cosf(th)) / (th * th);
    float K[9] = {0.f, -wz, wy, wz, 0.f, -wx, -wy, wx, 0.f};
    float K2[9];
    for (int r = 0; r < 3; ++r)
      for (int c = 0; c < 3; ++c) {
        float s = 0.f;
        for (int j = 0; j < 3; ++j) s += K[r * 3 + j] * K[j * 3 + c];
        K2[r * 3 + c] = s;
      }
    float* Ro = rts + t * 13;
    for (int i = 0; i < 9; ++i)
      Ro[i] = ((i % 4 == 0) ? 1.f : 0.f) + a * K[i] + bb * K2[i];
    Ro[9] = pp[3]; Ro[10] = pp[4]; Ro[11] = pp[5];
    Ro[12] = scale[t];
  }
}

// ---------------------------------------------------------------------------
// Main fused kernel: one WG = 128 points x 1 batch, 8 wave32s, ~920 WMMA/wave.
// ---------------------------------------------------------------------------
struct GHParams {
  const float* xyz0; const float* feature0; const float* scales0;
  const float* rotation0; const float* opacity0; const float* landmarks;
  const _Float16* w[16];
  const float* bias[16];
  const _Float16* expv; const _Float16* posev; const float* rts;
  float* out;
};

__global__ __launch_bounds__(256) void gh_main(GHParams P) {
  extern __shared__ char smem[];
  _Float16* bufA    = (_Float16*)(smem + SH_BUFA);
  _Float16* bufB    = (_Float16*)(smem + SH_BUFB);
  _Float16* s_feat  = (_Float16*)(smem + SH_FEAT);
  _Float16* s_xyzpe = (_Float16*)(smem + SH_XPE);
  float*    s_color = (float*)(smem + SH_COLOR);   // [128][36]
  float*    s_dattr = (float*)(smem + SH_DATTR);   // [128][9]
  float*    s_dxyz  = (float*)(smem + SH_DXYZ);    // [128][4]
  float*    s_ew    = (float*)(smem + SH_EW);
  float*    s_xyz   = (float*)(smem + SH_XYZ);     // [128][3]
  float*    s_lm    = (float*)(smem + SH_LM);      // [68][3]
  _Float16* s_expv  = (_Float16*)(smem + SH_VEC);
  _Float16* s_posev = s_expv + 64;
  float*    s_rts   = (float*)(smem + SH_RTS);

  const int b    = blockIdx.y;
  const int p0   = blockIdx.x * TM;
  const int tid  = threadIdx.x;
  const int lane = tid & 31;
  const int row0 = (tid >> 5) * 16;
  const int a4   = (lane >> 4) ? 4 : 0;
  const int b4   = (lane >> 4) ? 8 : 4;

  // ---- stage 0a: small shared data
  for (int i = tid; i < 68 * 3; i += 256) s_lm[i] = P.landmarks[i];
  if (tid < 64) {
    s_expv[tid]  = P.expv[b * 64 + tid];
    s_posev[tid] = P.posev[b * 64 + tid];
  }
  if (tid < 13) s_rts[tid] = P.rts[b * 13 + tid];
  __syncthreads();

  // ---- stage 0b: per-point features / posemb / ew
  {
    const int m  = tid & 127;
    const int h2 = tid >> 7;
    const int p  = p0 + m;
    const bool valid = p < NPTS;
    const float* src = P.feature0 + (size_t)p * 128 + h2 * 64;
    _Float16* dstf = s_feat + (size_t)m * FSTRIDE + h2 * 64;
#pragma unroll 4
    for (int i = 0; i < 16; ++i) {
      float4 f;
      if (valid) f = ((const float4*)src)[i];
      else { f.x = 0.f; f.y = 0.f; f.z = 0.f; f.w = 0.f; }
      dstf[i * 4 + 0] = (_Float16)tanhf(f.x);
      dstf[i * 4 + 1] = (_Float16)tanhf(f.y);
      dstf[i * 4 + 2] = (_Float16)tanhf(f.z);
      dstf[i * 4 + 3] = (_Float16)tanhf(f.w);
    }
    if (h2 == 0) {
      float x = 0.f, y = 0.f, z = 0.f;
      if (valid) {
        x = P.xyz0[(size_t)p * 3 + 0];
        y = P.xyz0[(size_t)p * 3 + 1];
        z = P.xyz0[(size_t)p * 3 + 2];
      }
      s_xyz[m * 3 + 0] = x; s_xyz[m * 3 + 1] = y; s_xyz[m * 3 + 2] = z;
      _Float16* xp = s_xyzpe + (size_t)m * 32;
      xp[0] = (_Float16)x; xp[1] = (_Float16)y; xp[2] = (_Float16)z;
#pragma unroll
      for (int i = 0; i < 4; ++i) {
        float f = (float)(1 << i);
        xp[3 + 6 * i + 0] = (_Float16)sinf(x * f);
        xp[3 + 6 * i + 1] = (_Float16)sinf(y * f);
        xp[3 + 6 * i + 2] = (_Float16)sinf(z * f);
        xp[3 + 6 * i + 3] = (_Float16)cosf(x * f);
        xp[3 + 6 * i + 4] = (_Float16)cosf(y * f);
        xp[3 + 6 * i + 5] = (_Float16)cosf(z * f);
      }
#pragma unroll
      for (int j = 27; j < 32; ++j) xp[j] = (_Float16)0.f;
      float d2min = 1e30f;
      for (int l = 0; l < 68; ++l) {
        float dx = x - s_lm[l * 3 + 0];
        float dy = y - s_lm[l * 3 + 1];
        float dz = z - s_lm[l * 3 + 2];
        d2min = fminf(d2min, dx * dx + dy * dy + dz * dz);
      }
      s_ew[m] = fminf(fmaxf((0.3f - d2min) * 5.0f, 0.f), 1.f);
    }
  }
  __syncthreads();

  // ---- MLP pipeline (wave-private rows: no barriers needed between layers)
  // exp_color: 192->256->256->32
  build_fv(bufA, s_feat, s_expv, row0, lane);
  run_layer<192, 256, true>(bufA, bufB, P.w[0], P.bias[0], row0, lane, a4, b4);
  run_layer<256, 256, true>(bufB, bufA, P.w[1], P.bias[1], row0, lane, a4, b4);
  run_final<256, 32, false, false>(bufA, s_color, 36, P.w[2], P.bias[2], s_ew, false, row0, lane, a4, b4);
  // pose_color: 182(192)->128->32
  build_fv(bufA, s_feat, s_posev, row0, lane);
  run_layer<192, 128, true>(bufA, bufB, P.w[3], P.bias[3], row0, lane, a4, b4);
  run_final<128, 32, false, true>(bufB, s_color, 36, P.w[4], P.bias[4], s_ew, true, row0, lane, a4, b4);
  // exp_attr: 192->256->256->8
  build_fv(bufA, s_feat, s_expv, row0, lane);
  run_layer<192, 256, true>(bufA, bufB, P.w[5], P.bias[5], row0, lane, a4, b4);
  run_layer<256, 256, true>(bufB, bufA, P.w[6], P.bias[6], row0, lane, a4, b4);
  run_final<256, 8, false, false>(bufA, s_dattr, 9, P.w[7], P.bias[7], s_ew, false, row0, lane, a4, b4);
  // pose_attr: 182(192)->128->8
  build_fv(bufA, s_feat, s_posev, row0, lane);
  run_layer<192, 128, true>(bufA, bufB, P.w[8], P.bias[8], row0, lane, a4, b4);
  run_final<128, 8, false, true>(bufB, s_dattr, 9, P.w[9], P.bias[9], s_ew, true, row0, lane, a4, b4);
  // exp_deform: 91(96)->256->256->256->3 (tanh)
  build_dv(bufA, s_xyzpe, s_expv, row0, lane);
  run_layer<96, 256, true>(bufA, bufB, P.w[10], P.bias[10], row0, lane, a4, b4);
  run_layer<256, 256, true>(bufB, bufA, P.w[11], P.bias[11], row0, lane, a4, b4);
  run_layer<256, 256, true>(bufA, bufB, P.w[12], P.bias[12], row0, lane, a4, b4);
  run_final<256, 3, true, false>(bufB, s_dxyz, 4, P.w[13], P.bias[13], s_ew, false, row0, lane, a4, b4);
  // pose_deform: 81(96)->128->3 (tanh)
  build_dv(bufA, s_xyzpe, s_posev, row0, lane);
  run_layer<96, 128, true>(bufA, bufB, P.w[14], P.bias[14], row0, lane, a4, b4);
  run_final<128, 3, true, true>(bufB, s_dxyz, 4, P.w[15], P.bias[15], s_ew, true, row0, lane, a4, b4);

  __syncthreads();

  // ---- epilogue: one thread per point
  if (tid < TM) {
    const int m = tid;
    const int p = p0 + m;
    if (p < NPTS) {
      const float R0 = s_rts[0], R1 = s_rts[1], R2 = s_rts[2];
      const float R3 = s_rts[3], R4 = s_rts[4], R5 = s_rts[5];
      const float R6 = s_rts[6], R7 = s_rts[7], R8 = s_rts[8];
      const float Tx = s_rts[9], Ty = s_rts[10], Tz = s_rts[11], Sc = s_rts[12];
      const size_t bp = (size_t)b * NPTS + p;
      // xyz = R @ ((xyz0 + dxyz*0.3) * S) + T
      float x = (s_xyz[m * 3 + 0] + s_dxyz[m * 4 + 0] * 0.3f) * Sc;
      float y = (s_xyz[m * 3 + 1] + s_dxyz[m * 4 + 1] * 0.3f) * Sc;
      float z = (s_xyz[m * 3 + 2] + s_dxyz[m * 4 + 2] * 0.3f) * Sc;
      P.out[OUT_XYZ + bp * 3 + 0] = R0 * x + R1 * y + R2 * z + Tx;
      P.out[OUT_XYZ + bp * 3 + 1] = R3 * x + R4 * y + R5 * z + Ty;
      P.out[OUT_XYZ + bp * 3 + 2] = R6 * x + R7 * y + R8 * z + Tz;
      // color
      for (int c = 0; c < 32; ++c)
        P.out[OUT_COLOR + bp * 32 + c] = s_color[m * 36 + c];
      // scales
      for (int j = 0; j < 3; ++j)
        P.out[OUT_SCALES + bp * 3 + j] =
            expf(P.scales0[(size_t)p * 3 + j] + s_dattr[m * 9 + j] * 0.05f) * Sc;
      // rot: normalize quat, quat->mat, M = R@Q, mat->quat
      float q0 = P.rotation0[(size_t)p * 4 + 0] + s_dattr[m * 9 + 3] * 0.05f;
      float q1 = P.rotation0[(size_t)p * 4 + 1] + s_dattr[m * 9 + 4] * 0.05f;
      float q2 = P.rotation0[(size_t)p * 4 + 2] + s_dattr[m * 9 + 5] * 0.05f;
      float q3 = P.rotation0[(size_t)p * 4 + 3] + s_dattr[m * 9 + 6] * 0.05f;
      float nrm = fmaxf(sqrtf(q0 * q0 + q1 * q1 + q2 * q2 + q3 * q3), 1e-12f);
      q0 /= nrm; q1 /= nrm; q2 /= nrm; q3 /= nrm;
      float two_s = 2.f / (q0 * q0 + q1 * q1 + q2 * q2 + q3 * q3);
      float r = q0, i = q1, j = q2, k = q3;
      float Q[9];
      Q[0] = 1.f - two_s * (j * j + k * k); Q[1] = two_s * (i * j - k * r); Q[2] = two_s * (i * k + j * r);
      Q[3] = two_s * (i * j + k * r); Q[4] = 1.f - two_s * (i * i + k * k); Q[5] = two_s * (j * k - i * r);
      Q[6] = two_s * (i * k - j * r); Q[7] = two_s * (j * k + i * r); Q[8] = 1.f - two_s * (i * i + j * j);
      float M[9];
      M[0] = R0 * Q[0] + R1 * Q[3] + R2 * Q[6];
      M[1] = R0 * Q[1] + R1 * Q[4] + R2 * Q[7];
      M[2] = R0 * Q[2] + R1 * Q[5] + R2 * Q[8];
      M[3] = R3 * Q[0] + R4 * Q[3] + R5 * Q[6];
      M[4] = R3 * Q[1] + R4 * Q[4] + R5 * Q[7];
      M[5] = R3 * Q[2] + R4 * Q[5] + R5 * Q[8];
      M[6] = R6 * Q[0] + R7 * Q[3] + R8 * Q[6];
      M[7] = R6 * Q[1] + R7 * Q[4] + R8 * Q[7];
      M[8] = R6 * Q[2] + R7 * Q[5] + R8 * Q[8];
      float t0 = 1.f + M[0] + M[4] + M[8];
      float t1 = 1.f + M[0] - M[4] - M[8];
      float t2 = 1.f - M[0] + M[4] - M[8];
      float t3 = 1.f - M[0] - M[4] + M[8];
      float qa0 = t0 > 0.f ? sqrtf(t0) : 0.f;
      float qa1 = t1 > 0.f ? sqrtf(t1) : 0.f;
      float qa2 = t2 > 0.f ? sqrtf(t2) : 0.f;
      float qa3 = t3 > 0.f ? sqrtf(t3) : 0.f;
      int idx = 0; float best = qa0;
      if (qa1 > best) { best = qa1; idx = 1; }
      if (qa2 > best) { best = qa2; idx = 2; }
      if (qa3 > best) { best = qa3; idx = 3; }
      float c0, c1, c2, c3;
      if (idx == 0)      { c0 = qa0 * qa0; c1 = M[7] - M[5]; c2 = M[2] - M[6]; c3 = M[3] - M[1]; }
      else if (idx == 1) { c0 = M[7] - M[5]; c1 = qa1 * qa1; c2 = M[3] + M[1]; c3 = M[2] + M[6]; }
      else if (idx == 2) { c0 = M[2] - M[6]; c1 = M[3] + M[1]; c2 = qa2 * qa2; c3 = M[5] + M[7]; }
      else               { c0 = M[3] - M[1]; c1 = M[6] + M[2]; c2 = M[7] + M[5]; c3 = qa3 * qa3; }
      float denom = 2.f * fmaxf(best, 0.1f);
      P.out[OUT_ROT + bp * 4 + 0] = c0 / denom;
      P.out[OUT_ROT + bp * 4 + 1] = c1 / denom;
      P.out[OUT_ROT + bp * 4 + 2] = c2 / denom;
      P.out[OUT_ROT + bp * 4 + 3] = c3 / denom;
      // opacity
      float o = P.opacity0[p] + s_dattr[m * 9 + 7] * 0.05f;
      P.out[OUT_OP + bp] = 1.f / (1.f + expf(-o));
    }
  }
}

// ---------------------------------------------------------------------------
// Host side
// ---------------------------------------------------------------------------
struct LayerDef { int widx; int nout; int k; int kp; int np; };
static const LayerDef LDEFS[16] = {
  { 8, 256, 192, 192, 256}, {10, 256, 256, 256, 256}, {12,  32, 256, 256,  32},  // exp_color
  {14, 128, 182, 192, 128}, {16,  32, 128, 128,  32},                            // pose_color
  {18, 256, 192, 192, 256}, {20, 256, 256, 256, 256}, {22,   8, 256, 256,  16},  // exp_attr
  {24, 128, 182, 192, 128}, {26,   8, 128, 128,  16},                            // pose_attr
  {28, 256,  91,  96, 256}, {30, 256, 256, 256, 256}, {32, 256, 256, 256, 256},
  {34,   3, 256, 256,  16},                                                      // exp_deform
  {36, 128,  81,  96, 128}, {38,   3, 128, 128,  16},                            // pose_deform
};

extern "C" void kernel_launch(void* const* d_in, const int* in_sizes, int n_in,
                              void* d_out, int out_size, void* d_ws, size_t ws_size,
                              hipStream_t stream) {
  (void)in_sizes; (void)n_in; (void)out_size; (void)ws_size;
  const float* exp_coeff = (const float*)d_in[0];
  const float* pose      = (const float*)d_in[1];
  const float* scale     = (const float*)d_in[2];

  _Float16* wbase = (_Float16*)d_ws;
  size_t off[16], cur = 0;
  for (int i = 0; i < 16; ++i) { off[i] = cur; cur += (size_t)LDEFS[i].kp * LDEFS[i].np; }
  _Float16* expv  = wbase + cur;          // 128 f16
  _Float16* posev = expv + BATCH * 64;    // 128 f16
  float*    rts   = (float*)(posev + BATCH * 64);  // 26 f32  (ws use < 1 MB total)

  // pack weights into WMMA fragment order (f16)
  for (int i = 0; i < 16; ++i) {
    int total = (LDEFS[i].np / 16) * (LDEFS[i].kp / 32) * 256;
    int grid = (total + 255) / 256;
    gh_pack<<<grid, 256, 0, stream>>>((const float*)d_in[LDEFS[i].widx], wbase + off[i],
                                      LDEFS[i].nout, LDEFS[i].k, LDEFS[i].kp, LDEFS[i].np);
  }
  gh_prep<<<1, 128, 0, stream>>>(exp_coeff, pose, scale, expv, posev, rts);

  GHParams P;
  P.xyz0      = (const float*)d_in[3];
  P.feature0  = (const float*)d_in[4];
  P.scales0   = (const float*)d_in[5];
  P.rotation0 = (const float*)d_in[6];
  P.opacity0  = (const float*)d_in[7];
  P.landmarks = (const float*)d_in[40];
  for (int i = 0; i < 16; ++i) {
    P.w[i]    = wbase + off[i];
    P.bias[i] = (const float*)d_in[LDEFS[i].widx + 1];
  }
  P.expv = expv; P.posev = posev; P.rts = rts;
  P.out = (float*)d_out;

  hipFuncSetAttribute(reinterpret_cast<const void*>(&gh_main),
                      hipFuncAttributeMaxDynamicSharedMemorySize, (int)SH_TOTAL);
  gh_main<<<dim3(TILES, BATCH), 256, SH_TOTAL, stream>>>(P);
}
